// VisualAttention_6760278524500
// MI455X (gfx1250) — compile-verified, matrix operand order checked by reading, not run on hardware
//
#include <hip/hip_runtime.h>
#include <hip/hip_bf16.h>

typedef __bf16 bf16_t;
typedef __attribute__((ext_vector_type(16))) __bf16 v16bf;
typedef __attribute__((ext_vector_type(8)))  __bf16 v8bf;
typedef __attribute__((ext_vector_type(8)))  float  v8f;

constexpr int kSeq   = 1024;
constexpr int kBatch = 8;
constexpr int kEmbed = 1024;
constexpr int kHeads = 16;
constexpr int kHd    = 64;
constexpr int kRows  = kSeq * kBatch;   // 8192
constexpr int kE3    = 3 * kEmbed;      // 3072
constexpr int kPitch = 40;              // padded LDS row pitch (elems)

// ---------------------------------------------------------------------------
// helpers
// ---------------------------------------------------------------------------
__device__ __forceinline__ v8f vzero8() {
    v8f z = {0.f, 0.f, 0.f, 0.f, 0.f, 0.f, 0.f, 0.f};
    return z;
}

// Load one 16x32 bf16 WMMA operand fragment from global memory.
// "Major" index (rows for A, columns for B) has stride `stride`, K contiguous.
// CDNA5 16-bit layout: lanes 0-15 hold K[0..7]+[16..23], lanes 16-31 hold
// K[8..15]+[24..31], for major index (lane&15).
__device__ __forceinline__ v16bf load_frag(const bf16_t* __restrict__ p, int stride) {
    const int lane = threadIdx.x & 31;
    const bf16_t* b = p + (size_t)(lane & 15) * stride + (lane >> 4) * 8;
    v8bf lo = *reinterpret_cast<const v8bf*>(b);
    v8bf hi = *reinterpret_cast<const v8bf*>(b + 16);
    return __builtin_shufflevector(lo, hi, 0, 1, 2, 3, 4, 5, 6, 7,
                                           8, 9, 10, 11, 12, 13, 14, 15);
}

// Same fragment pattern from an LDS tile with row pitch kPitch.
__device__ __forceinline__ v16bf lds_frag(const bf16_t* p) {
    const int lane = threadIdx.x & 31;
    const bf16_t* b = p + (lane & 15) * kPitch + (lane >> 4) * 8;
    v8bf lo = *reinterpret_cast<const v8bf*>(b);
    v8bf hi = *reinterpret_cast<const v8bf*>(b + 16);
    return __builtin_shufflevector(lo, hi, 0, 1, 2, 3, 4, 5, 6, 7,
                                           8, 9, 10, 11, 12, 13, 14, 15);
}

__device__ __forceinline__ v8f wmma_bf16(v16bf a, v16bf b, v8f c) {
    return __builtin_amdgcn_wmma_f32_16x16x32_bf16(
        /*neg_a=*/false, a, /*neg_b=*/false, b,
        /*c_mod=*/(short)0, c, /*reuse_a=*/false, /*reuse_b=*/false);
}

// Async 16B global -> LDS copy. The LDS byte offset is the low 32 bits of the
// flat address of the shared-memory destination; deriving it via ptrtoint
// escapes the LDS object so the "memory" clobber aliases it (keeps the
// subsequent ds_load fragments alive).
__device__ __forceinline__ void async_b128(const bf16_t* g, const bf16_t* l) {
    const uint32_t loff = (uint32_t)(uintptr_t)l;
    asm volatile("global_load_async_to_lds_b128 %0, %1, off"
                 :: "v"(loff), "v"(g) : "memory");
}

#define ASYNC_WAIT_4() asm volatile("s_wait_asynccnt 0x4" ::: "memory")
#define ASYNC_WAIT_0() asm volatile("s_wait_asynccnt 0x0" ::: "memory")

// ---------------------------------------------------------------------------
// f32 -> bf16 bulk convert (8 elems / thread, 16B stores)
// ---------------------------------------------------------------------------
__global__ void k_f32_to_bf16(const float* __restrict__ in,
                              bf16_t* __restrict__ out, int n8) {
    int i = blockIdx.x * blockDim.x + threadIdx.x;
    if (i >= n8) return;
    const float4* p = reinterpret_cast<const float4*>(in) + (size_t)i * 2;
    float4 a = p[0];
    float4 b = p[1];
    v8bf o;
    o[0] = (bf16_t)a.x; o[1] = (bf16_t)a.y; o[2] = (bf16_t)a.z; o[3] = (bf16_t)a.w;
    o[4] = (bf16_t)b.x; o[5] = (bf16_t)b.y; o[6] = (bf16_t)b.z; o[7] = (bf16_t)b.w;
    reinterpret_cast<v8bf*>(out)[i] = o;
}

// ---------------------------------------------------------------------------
// Async-staged, double-buffered GEMM main loop (block tile 128x128, K=1024).
// A: row-major [M,1024]; B: K-contiguous per output column [N,1024].
// smem[buf][0] = A tile 128x32, smem[buf][1] = B tile 128x32 (pitch 40).
// Each of the 256 threads moves 2x16B per tile per chunk via
// GLOBAL_LOAD_ASYNC_TO_LDS_B128 (ASYNCcnt-tracked), then waves re-read
// fragments with ds_load_b128. Explicit 2-chunk unroll keeps buffer
// indices compile-time constant.
// ---------------------------------------------------------------------------
__device__ __forceinline__ void gemm_lds_mainloop(
    const bf16_t* __restrict__ A, const bf16_t* __restrict__ B,
    int m0blk, int n0blk,
    bf16_t (&smem)[2][2][128][kPitch], v8f (&acc)[4][2]) {
    const int tid  = threadIdx.x;
    const int wave = tid >> 5;
    const int wm   = wave >> 2;   // 0..1
    const int wn   = wave & 3;    // 0..3
    const int crow = tid >> 1;    // 0..127: tile row this thread copies
    const int cseg = (tid & 1) * 16;  // elem offset within 32-elem row

    const bf16_t* gA = A + (size_t)(m0blk + crow) * kEmbed + cseg;
    const bf16_t* gB = B + (size_t)(n0blk + crow) * kEmbed + cseg;

    auto issue = [&](int k, int buf) {
        const bf16_t* ga = gA + k;
        const bf16_t* gb = gB + k;
        bf16_t* la = &smem[buf][0][crow][cseg];
        bf16_t* lb = &smem[buf][1][crow][cseg];
        async_b128(ga,     la);
        async_b128(ga + 8, la + 8);
        async_b128(gb,     lb);
        async_b128(gb + 8, lb + 8);
    };

    auto compute = [&](int buf) {
        v16bf af[4];
#pragma unroll
        for (int mi = 0; mi < 4; mi++)
            af[mi] = lds_frag(&smem[buf][0][wm * 64 + mi * 16][0]);
        v16bf bf2[2];
#pragma unroll
        for (int ni = 0; ni < 2; ni++)
            bf2[ni] = lds_frag(&smem[buf][1][wn * 32 + ni * 16][0]);
#pragma unroll
        for (int mi = 0; mi < 4; mi++)
#pragma unroll
            for (int ni = 0; ni < 2; ni++)
                acc[mi][ni] = wmma_bf16(af[mi], bf2[ni], acc[mi][ni]);
    };

    constexpr int kChunks = kEmbed / 32;  // 32 (even)
    issue(0, 0);
#pragma unroll 1
    for (int c = 0; c < kChunks; c += 2) {
        // ---- even chunk (buffer 0) ----
        issue((c + 1) * 32, 1);           // c+1 < kChunks always (c <= 30)
        ASYNC_WAIT_4();                   // chunk c's 4 async ops retired
        __syncthreads();
        compute(0);
        __syncthreads();
        // ---- odd chunk (buffer 1) ----
        if (c + 2 < kChunks) {
            issue((c + 2) * 32, 0);
            ASYNC_WAIT_4();
        } else {
            ASYNC_WAIT_0();
        }
        __syncthreads();
        compute(1);
        __syncthreads();
    }
}

// ---------------------------------------------------------------------------
// Fused QKV projection: mixed = X @ W^T + bias, scatter into q/k/v buffers.
// q,k layout: [n=b*16+h][s][64] (bf16); q pre-scaled by 1/8.
// v layout:   [n][d][s] (bf16, K-contiguous in s for the ctx GEMM).
// ---------------------------------------------------------------------------
__global__ __launch_bounds__(256)
void k_gemm_qkv(const bf16_t* __restrict__ X, const bf16_t* __restrict__ W,
                const float* __restrict__ bias,
                bf16_t* __restrict__ qb, bf16_t* __restrict__ kb,
                bf16_t* __restrict__ vb) {
    __shared__ __align__(16) bf16_t smem[2][2][128][kPitch];
    const int wave = threadIdx.x >> 5;
    const int wm = wave >> 2;
    const int wn = wave & 3;
    const int m0 = blockIdx.x * 128 + wm * 64;
    const int n0 = blockIdx.y * 128 + wn * 32;

    v8f acc[4][2];
#pragma unroll
    for (int mi = 0; mi < 4; mi++)
#pragma unroll
        for (int ni = 0; ni < 2; ni++) acc[mi][ni] = vzero8();

    gemm_lds_mainloop(X, W, blockIdx.x * 128, blockIdx.y * 128, smem, acc);

    const int lane = threadIdx.x & 31;
    const int col  = lane & 15;
    const int half = lane >> 4;
#pragma unroll
    for (int ni = 0; ni < 2; ni++) {
        const int c   = n0 + ni * 16 + col;   // 0..3071
        const int h   = c / 192;
        const int rem = c - h * 192;
        const int t   = rem >> 6;             // 0=q 1=k 2=v
        const int d   = rem & 63;
        const float bv = bias[c];
#pragma unroll
        for (int mi = 0; mi < 4; mi++) {
#pragma unroll
            for (int i = 0; i < 8; i++) {
                const int r  = m0 + mi * 16 + i + half * 8;
                const int s  = r >> 3;        // seq index
                const int b  = r & 7;         // batch index
                const int nh = b * kHeads + h;
                const float v = acc[mi][ni][i] + bv;
                if (t == 0)
                    qb[((size_t)nh * kSeq + s) * kHd + d] = (bf16_t)(v * 0.125f);
                else if (t == 1)
                    kb[((size_t)nh * kSeq + s) * kHd + d] = (bf16_t)v;
                else
                    vb[((size_t)nh * kHd + d) * kSeq + s] = (bf16_t)v;
            }
        }
    }
}

// ---------------------------------------------------------------------------
// Flash-attention core: per block = 1 head x 64 q rows, 4 waves x 16 rows.
// Streams kv in chunks of 32 with online softmax. Writes ctx (bf16) in
// out-projection layout: ctx[(s*B + b)*E + h*64 + d].
// ---------------------------------------------------------------------------
__global__ __launch_bounds__(128)
void k_attn(const bf16_t* __restrict__ qb, const bf16_t* __restrict__ kb,
            const bf16_t* __restrict__ vb, bf16_t* __restrict__ ctx) {
    __shared__ __align__(16) bf16_t lds[4][16][kPitch];  // padded P tile per wave

    const int wave = threadIdx.x >> 5;
    const int lane = threadIdx.x & 31;
    const int col  = lane & 15;
    const int half = lane >> 4;

    const int n = blockIdx.y;            // head slot = b*16 + h
    const int b = n >> 4;
    const int h = n & 15;
    const int q0 = blockIdx.x * 64 + wave * 16;

    const bf16_t* qh = qb + (size_t)n * kSeq * kHd;
    const bf16_t* kh = kb + (size_t)n * kSeq * kHd;
    const bf16_t* vh = vb + (size_t)n * kHd * kSeq;

    // Q fragments for this wave's 16 rows (K = d, split 0..31 / 32..63)
    const v16bf qf0 = load_frag(qh + (size_t)q0 * kHd, kHd);
    const v16bf qf1 = load_frag(qh + (size_t)q0 * kHd + 32, kHd);

    v8f accO[4];
    float mrow[8], lrow[8];
#pragma unroll
    for (int t = 0; t < 4; t++) accO[t] = vzero8();
#pragma unroll
    for (int i = 0; i < 8; i++) { mrow[i] = -1e30f; lrow[i] = 0.f; }

    for (int kc = 0; kc < kSeq; kc += 32) {
        // K^T fragments: columns are k-indices, K-dim is d
        v16bf k0a = load_frag(kh + (size_t)kc * kHd, kHd);
        v16bf k0b = load_frag(kh + (size_t)kc * kHd + 32, kHd);
        v16bf k1a = load_frag(kh + (size_t)(kc + 16) * kHd, kHd);
        v16bf k1b = load_frag(kh + (size_t)(kc + 16) * kHd + 32, kHd);

        v8f S0 = vzero8();
        S0 = wmma_bf16(qf0, k0a, S0);
        S0 = wmma_bf16(qf1, k0b, S0);
        v8f S1 = vzero8();
        S1 = wmma_bf16(qf0, k1a, S1);
        S1 = wmma_bf16(qf1, k1b, S1);

        // row max over 32 columns (2 per lane, then 16-lane butterfly)
        float rm[8];
#pragma unroll
        for (int i = 0; i < 8; i++) rm[i] = fmaxf(S0[i], S1[i]);
#pragma unroll
        for (int msk = 1; msk < 16; msk <<= 1)
#pragma unroll
            for (int i = 0; i < 8; i++)
                rm[i] = fmaxf(rm[i], __shfl_xor(rm[i], msk, 32));

        float rs[8];
#pragma unroll
        for (int i = 0; i < 8; i++) {
            const float mn = fmaxf(mrow[i], rm[i]);
            const float sc = __expf(mrow[i] - mn);
            mrow[i] = mn;
            const float p0 = __expf(S0[i] - mn);
            const float p1 = __expf(S1[i] - mn);
            const int row = i + half * 8;
            lds[wave][row][col]      = (bf16_t)p0;
            lds[wave][row][16 + col] = (bf16_t)p1;
            rs[i] = p0 + p1;
            lrow[i] *= sc;
#pragma unroll
            for (int t = 0; t < 4; t++) accO[t][i] *= sc;
        }
#pragma unroll
        for (int msk = 1; msk < 16; msk <<= 1)
#pragma unroll
            for (int i = 0; i < 8; i++) rs[i] += __shfl_xor(rs[i], msk, 32);
#pragma unroll
        for (int i = 0; i < 8; i++) lrow[i] += rs[i];

        // Re-fragment P from LDS into WMMA A layout (rows=q, K=k chunk)
        v16bf pf = lds_frag(&lds[wave][0][0]);

        // ctx += P @ V   (V: columns are d, K-dim is k — contiguous in s)
#pragma unroll
        for (int t = 0; t < 4; t++) {
            v16bf vf = load_frag(vh + (size_t)(t * 16) * kSeq + kc, kSeq);
            accO[t] = wmma_bf16(pf, vf, accO[t]);
        }
    }

    // normalize and store bf16 ctx in out-proj GEMM layout
#pragma unroll
    for (int i = 0; i < 8; i++) {
        const float inv = 1.0f / lrow[i];
        const int s = q0 + i + half * 8;
#pragma unroll
        for (int t = 0; t < 4; t++) {
            const int e = h * kHd + t * 16 + col;
            ctx[((size_t)s * kBatch + b) * kEmbed + e] = (bf16_t)(accO[t][i] * inv);
        }
    }
}

// ---------------------------------------------------------------------------
// Output projection: out = ctx @ Wout^T + bias (fp32 result)
// ---------------------------------------------------------------------------
__global__ __launch_bounds__(256)
void k_gemm_out(const bf16_t* __restrict__ X, const bf16_t* __restrict__ W,
                const float* __restrict__ bias, float* __restrict__ out) {
    __shared__ __align__(16) bf16_t smem[2][2][128][kPitch];
    const int wave = threadIdx.x >> 5;
    const int wm = wave >> 2;
    const int wn = wave & 3;
    const int m0 = blockIdx.x * 128 + wm * 64;
    const int n0 = blockIdx.y * 128 + wn * 32;

    v8f acc[4][2];
#pragma unroll
    for (int mi = 0; mi < 4; mi++)
#pragma unroll
        for (int ni = 0; ni < 2; ni++) acc[mi][ni] = vzero8();

    gemm_lds_mainloop(X, W, blockIdx.x * 128, blockIdx.y * 128, smem, acc);

    const int lane = threadIdx.x & 31;
    const int col  = lane & 15;
    const int half = lane >> 4;
#pragma unroll
    for (int ni = 0; ni < 2; ni++) {
        const int c = n0 + ni * 16 + col;
        const float bv = bias[c];
#pragma unroll
        for (int mi = 0; mi < 4; mi++) {
#pragma unroll
            for (int i = 0; i < 8; i++) {
                const int r = m0 + mi * 16 + i + half * 8;
                out[(size_t)r * kEmbed + c] = acc[mi][ni][i] + bv;
            }
        }
    }
}

// ---------------------------------------------------------------------------
// launch
// ---------------------------------------------------------------------------
extern "C" void kernel_launch(void* const* d_in, const int* in_sizes, int n_in,
                              void* d_out, int out_size, void* d_ws, size_t ws_size,
                              hipStream_t stream) {
    const float* query      = (const float*)d_in[0];
    const float* in_proj_w  = (const float*)d_in[3];
    const float* in_proj_b  = (const float*)d_in[4];
    const float* out_proj_w = (const float*)d_in[5];
    const float* out_proj_b = (const float*)d_in[6];
    float* out = (float*)d_out;

    char* ws = (char*)d_ws;
    const size_t szRowsE  = (size_t)kRows * kEmbed * sizeof(bf16_t);   // 16 MB
    const size_t szWqkv   = (size_t)kE3 * kEmbed * sizeof(bf16_t);     //  6 MB
    const size_t szWout   = (size_t)kEmbed * kEmbed * sizeof(bf16_t);  //  2 MB
    const size_t szHeadQK = (size_t)kBatch * kHeads * kSeq * kHd * sizeof(bf16_t); // 16 MB

    bf16_t* xact = (bf16_t*)(ws);                               // converted query
    bf16_t* wqkv = (bf16_t*)(ws + szRowsE);                     // converted in_proj_w
    bf16_t* wout = (bf16_t*)(ws + szRowsE + szWqkv);            // converted out_proj_w
    bf16_t* qbuf = (bf16_t*)(ws + szRowsE + szWqkv + szWout);
    bf16_t* kbuf = (bf16_t*)((char*)qbuf + szHeadQK);
    bf16_t* vbuf = (bf16_t*)((char*)kbuf + szHeadQK);
    bf16_t* ctx2 = (bf16_t*)((char*)vbuf + szHeadQK);

    // 1) convert fp32 -> bf16
    {
        int n8 = kRows * kEmbed / 8;
        k_f32_to_bf16<<<(n8 + 255) / 256, 256, 0, stream>>>(query, xact, n8);
        n8 = kE3 * kEmbed / 8;
        k_f32_to_bf16<<<(n8 + 255) / 256, 256, 0, stream>>>(in_proj_w, wqkv, n8);
        n8 = kEmbed * kEmbed / 8;
        k_f32_to_bf16<<<(n8 + 255) / 256, 256, 0, stream>>>(out_proj_w, wout, n8);
    }

    // 2) fused QKV projection + head scatter (async-LDS staged GEMM)
    k_gemm_qkv<<<dim3(kRows / 128, kE3 / 128), 256, 0, stream>>>(
        xact, wqkv, in_proj_b, qbuf, kbuf, vbuf);

    // 3) streaming attention (flash-style, online softmax)
    k_attn<<<dim3(kSeq / 64, kBatch * kHeads), 128, 0, stream>>>(
        qbuf, kbuf, vbuf, ctx2);

    // 4) output projection (fp32 result, async-LDS staged GEMM)
    k_gemm_out<<<dim3(kRows / 128, kEmbed / 128), 256, 0, stream>>>(
        ctx2, wout, out_proj_b, out);
}
